// LSTMEncoder_33715493273744
// MI455X (gfx1250) — matrix operand order repaired and checked by reading, not compiled, not run on hardware
//
#include <hip/hip_runtime.h>

// ---------------------------------------------------------------------------
// LSTM encoder for MI455X (gfx1250): bf16 WMMA GEMMs + batch-split serial scan
// ---------------------------------------------------------------------------

typedef __bf16 bf16_t;
typedef __attribute__((ext_vector_type(16))) __bf16 v16bf;
typedef __attribute__((ext_vector_type(8)))  __bf16 v8bf;
typedef __attribute__((ext_vector_type(8)))  float  v8f;

// explicit global (AS1) pointer types so laundered pointers still emit
// global_load_* (not flat_load_*, which ties up DScnt + the LDS pipe)
typedef const bf16_t __attribute__((address_space(1)))* gbf16p;
typedef const v8bf   __attribute__((address_space(1)))* gv8bfp;

#define B_  64
#define T_  2048
#define X_  128
#define H_  256
#define G_  1024            // 4*H
#define Z_  64
#define S_  (B_ * T_)       // 131072 samples

union V16 { v16bf v; v8bf h[2]; };

// A fragment (16x32 bf16, MxK), A row-major with leading dim lda.
// ISA 7.12.2: lane<16 -> row=lane,   elems 0..7 = K k0..k0+7,  8..15 = K k0+16..23
//             lane>=16 -> row=lane-16, elems 0..7 = K k0+8..15, 8..15 = K k0+24..31
__device__ __forceinline__ v16bf load_frag_A(const bf16_t* A, int lda,
                                             int m0, int k0, int lane) {
  int row = m0 + (lane & 15);
  int kb  = k0 + ((lane >> 4) << 3);
  const bf16_t* p = A + (size_t)row * lda + kb;
  V16 r;
  r.h[0] = *(const v8bf*)(p);
  r.h[1] = *(const v8bf*)(p + 16);
  return r.v;
}

// B fragment (32x16 bf16, KxN), weights stored transposed: BT is N x K row-major.
// lane<16 -> col=lane, elems 0..7 = K k0..k0+7, 8..15 = K k0+16..23 ; lane>=16 -> col=lane-16, K k0+8..15, k0+24..31
__device__ __forceinline__ v16bf load_frag_B(const bf16_t* BT, int ldb,
                                             int n0, int k0, int lane) {
  int col = n0 + (lane & 15);
  int kb  = k0 + ((lane >> 4) << 3);
  const bf16_t* p = BT + (size_t)col * ldb + kb;
  V16 r;
  r.h[0] = *(const v8bf*)(p);
  r.h[1] = *(const v8bf*)(p + 16);
  return r.v;
}

// Same as load_frag_B but through an explicit global-address-space pointer.
__device__ __forceinline__ v16bf load_frag_B_g(gbf16p BT, int ldb,
                                               int n0, int k0, int lane) {
  int col = n0 + (lane & 15);
  int kb  = k0 + ((lane >> 4) << 3);
  gbf16p p = BT + (size_t)col * ldb + kb;
  V16 r;
  r.h[0] = *(gv8bfp)(p);
  r.h[1] = *(gv8bfp)(p + 16);
  return r.v;
}

__device__ __forceinline__ v8f wmma_bf16(v16bf a, v16bf b, v8f c) {
  return __builtin_amdgcn_wmma_f32_16x16x32_bf16(false, a, false, b, (short)0, c,
                                                 false, false);
}

__device__ __forceinline__ float sigf(float x) {
  return 1.0f / (1.0f + __expf(-x));
}
__device__ __forceinline__ float tanhfast(float x) {
  float e = __expf(-2.0f * x);
  return (1.0f - e) / (1.0f + e);
}

// ---------------------------------------------------------------------------
// src: K x N row-major f32  ->  dst: N x K row-major bf16 (transpose+convert)
// ---------------------------------------------------------------------------
__global__ void transpose_to_bf16_kernel(const float* __restrict__ src,
                                         bf16_t* __restrict__ dst, int K, int N) {
  int idx = blockIdx.x * blockDim.x + threadIdx.x;
  if (idx >= K * N) return;
  int n = idx / K, k = idx - n * K;
  dst[idx] = (bf16_t)src[(size_t)k * N + n];
}

__global__ void f32_to_bf16_kernel(const float* __restrict__ src,
                                   bf16_t* __restrict__ dst, long long n) {
  long long i = (long long)blockIdx.x * blockDim.x + threadIdx.x;
  long long stride = (long long)gridDim.x * blockDim.x;
  for (; i < n; i += stride) dst[i] = (bf16_t)src[i];
}

// ---------------------------------------------------------------------------
// Generic WMMA GEMM: C(M x N) = A(M x K, bf16) * BT(N x K, bf16)^T [+ bias]
// Each wave computes one 16(M) x 64(N) strip: 4 accumulators, A frags reused.
// MODE 0: store f32 to Cf, remapped to (T, B, 4H) layout for the scan
// MODE 1: +bias, relu, store bf16 to Cb (hid)
// MODE 2: +bias, split-store f32: cols [0,64) -> Cf (mu), [64,128) -> Cf+S*64
// ---------------------------------------------------------------------------
template <int MODE, int N, int K>
__global__ __launch_bounds__(256) void wmma_gemm_kernel(
    const bf16_t* __restrict__ A, const bf16_t* __restrict__ BT,
    const float* __restrict__ bias, float* __restrict__ Cf,
    bf16_t* __restrict__ Cb) {
  const int lane = threadIdx.x & 31;
  const int wave = blockIdx.x * (blockDim.x >> 5) + (threadIdx.x >> 5);
  const int ngroups = N / 64;
  const int m0 = (wave / ngroups) * 16;
  const int n0 = (wave % ngroups) * 64;
  const int colA = lane & 15;
  const int rowm = (lane >> 4) << 3;

  v8f acc[4];
#pragma unroll
  for (int j = 0; j < 4; ++j) {
    float bv = (MODE == 0) ? 0.0f : bias[n0 + j * 16 + colA];
#pragma unroll
    for (int v = 0; v < 8; ++v) acc[j][v] = bv;
  }

#pragma unroll
  for (int kk = 0; kk < K / 32; ++kk) {
    v16bf af = load_frag_A(A, K, m0, kk * 32, lane);
#pragma unroll
    for (int j = 0; j < 4; ++j) {
      v16bf bf = load_frag_B(BT, K, n0 + j * 16, kk * 32, lane);
      acc[j] = wmma_bf16(af, bf, acc[j]);
    }
  }

#pragma unroll
  for (int j = 0; j < 4; ++j) {
#pragma unroll
    for (int v = 0; v < 8; ++v) {
      int row = m0 + rowm + v;   // sample index s = b*T + t
      int col = n0 + j * 16 + colA;
      float val = acc[j][v];
      if (MODE == 0) {
        // remap to (T, B, 4H): contiguous 64KB block per timestep for the scan
        int b = row >> 11;            // row / T
        int t = row & (T_ - 1);       // row % T
        Cf[((size_t)t * B_ + b) * N + col] = val;
      } else if (MODE == 1) {
        Cb[(size_t)row * N + col] = (bf16_t)fmaxf(val, 0.0f);
      } else {
        if (col < Z_)
          Cf[(size_t)row * Z_ + col] = val;                          // mu
        else
          Cf[(size_t)S_ * Z_ + (size_t)row * Z_ + (col - Z_)] = val; // log_sigma
      }
    }
  }
}

// ---------------------------------------------------------------------------
// Serial LSTM scan. Batches are independent rows of h@Wh, so split B=64 into
// 4 workgroups x 16 rows: zero cross-WGP sync on the serial path. 16 waves;
// wave w owns hidden-unit group u0=16w (its i/f/g/o gate columns) and keeps
// its c fragment in registers across all T steps. h is double-buffered in LDS
// (bf16) as the A operand for the next step's WMMA; one barrier per step.
//
// Register budget (~256 VGPRs @ 512 thr): i/f gate B-fragments (128 VGPRs) are
// pinned in registers for the whole scan; g/o gate B-fragments are streamed
// from L2 each step through a laundered GLOBAL-address-space pointer so the
// compiler can neither hoist them (scratch spills) nor demote them to flat
// loads (which would contend with the ds_load pipe / DScnt).
// ---------------------------------------------------------------------------
__global__ __launch_bounds__(512) void lstm_scan_kernel(
    const float* __restrict__ x_proj,  // (T, B, 4H) f32
    const bf16_t* __restrict__ WhT,    // (4H, H) bf16 (transposed Wh)
    const float* __restrict__ bh,      // (4H,)
    bf16_t* __restrict__ h_all) {      // (S, H) bf16 out, rows = b*T + t
  __shared__ bf16_t h_lds[2][16 * H_];

  const int tid  = threadIdx.x;
  const int lane = tid & 31;
  const int w    = tid >> 5;        // 0..15
  const int r0   = blockIdx.x * 16; // this block's batch rows
  const int u0   = w * 16;          // hidden unit group
  const int colA = lane & 15;
  const int rowm = (lane >> 4) << 3;

  for (int i = tid; i < 16 * H_; i += 512) h_lds[0][i] = (bf16_t)0.0f;
  __syncthreads();

  // Pin i/f gate B fragments in registers for the entire scan (16 x 8 VGPRs).
  v16bf Bi[8], Bf[8];
#pragma unroll
  for (int kk = 0; kk < 8; ++kk) {
    Bi[kk] = load_frag_B(WhT, H_, 0 * H_ + u0, kk * 32, lane);
    Bf[kk] = load_frag_B(WhT, H_, 1 * H_ + u0, kk * 32, lane);
  }

  v8f c_acc;
#pragma unroll
  for (int v = 0; v < 8; ++v) c_acc[v] = 0.0f;

  const float bias_i = bh[0 * H_ + u0 + colA];
  const float bias_f = bh[1 * H_ + u0 + colA];
  const float bias_g = bh[2 * H_ + u0 + colA];
  const float bias_o = bh[3 * H_ + u0 + colA];

  for (int t = 0; t < T_; ++t) {
    const int cur = t & 1, nxt = cur ^ 1;

    // Launder WhT (value-opaque per iteration) into a GLOBAL AS pointer:
    // g/o B fragments are re-issued as global_load_b128 from L2 every step.
    unsigned long long wo = (unsigned long long)WhT;
    asm volatile("" : "+s"(wo));
    gbf16p WhT_o = (gbf16p)wo;

    // Prefetch next timestep's x_proj block (16 x 4H f32 = 64KB, contiguous):
    // one 128B line per thread.
    if (t + 1 < T_) {
      const char* nxt_xp =
          (const char*)(x_proj + ((size_t)(t + 1) * B_ + r0) * G_);
      __builtin_prefetch(nxt_xp + (size_t)tid * 128, 0, 0);
    }

    // Accumulator init: x_proj + bias. x_proj is (T, B, 4H).
    const float* xp_t = x_proj + ((size_t)t * B_ + r0) * G_;
    v8f acc[4];
#pragma unroll
    for (int v = 0; v < 8; ++v) {
      const float* xr = xp_t + (size_t)(rowm + v) * G_;
      acc[0][v] = xr[0 * H_ + u0 + colA] + bias_i;
      acc[1][v] = xr[1 * H_ + u0 + colA] + bias_f;
      acc[2][v] = xr[2 * H_ + u0 + colA] + bias_g;
      acc[3][v] = xr[3 * H_ + u0 + colA] + bias_o;
    }

#pragma unroll
    for (int kk = 0; kk < H_ / 32; ++kk) {
      v16bf af = load_frag_A(&h_lds[cur][0], H_, 0, kk * 32, lane);
      acc[0] = wmma_bf16(af, Bi[kk], acc[0]);
      acc[1] = wmma_bf16(af, Bf[kk], acc[1]);
      v16bf bg = load_frag_B_g(WhT_o, H_, 2 * H_ + u0, kk * 32, lane);
      acc[2] = wmma_bf16(af, bg, acc[2]);
      v16bf bo = load_frag_B_g(WhT_o, H_, 3 * H_ + u0, kk * 32, lane);
      acc[3] = wmma_bf16(af, bo, acc[3]);
    }

#pragma unroll
    for (int v = 0; v < 8; ++v) {
      float ig = sigf(acc[0][v]);
      float fg = sigf(acc[1][v]);
      float gg = tanhfast(acc[2][v]);
      float og = sigf(acc[3][v]);
      float c  = fg * c_acc[v] + ig * gg;
      c_acc[v] = c;
      float h  = og * tanhfast(c);
      bf16_t hb = (bf16_t)h;
      int m = rowm + v;
      h_lds[nxt][m * H_ + (u0 + colA)] = hb;
      int b = r0 + m;
      h_all[((size_t)b * T_ + t) * H_ + (u0 + colA)] = hb;
    }
    __syncthreads();
  }
}

// ---------------------------------------------------------------------------
// Host orchestration
// ---------------------------------------------------------------------------
extern "C" void kernel_launch(void* const* d_in, const int* in_sizes, int n_in,
                              void* d_out, int out_size, void* d_ws,
                              size_t ws_size, hipStream_t stream) {
  const float* x  = (const float*)d_in[0];  // (B,T,X)
  const float* Wi = (const float*)d_in[1];  // (X,4H)
  const float* Wh = (const float*)d_in[2];  // (H,4H)
  const float* bh = (const float*)d_in[3];  // (4H,)
  const float* W1 = (const float*)d_in[4];  // (H,H)
  const float* b1 = (const float*)d_in[5];  // (H,)
  const float* W2 = (const float*)d_in[6];  // (H,2Z)
  const float* b2 = (const float*)d_in[7];  // (2Z,)
  float* out = (float*)d_out;               // mu (B,T,Z) ++ log_sigma (B,T,Z)

  char* ws = (char*)d_ws;
  size_t off = 0;
  auto walloc = [&](size_t bytes) -> void* {
    void* p = ws + off;
    off = (off + bytes + 255) & ~(size_t)255;
    return p;
  };
  bf16_t* WiT    = (bf16_t*)walloc((size_t)G_ * X_ * sizeof(bf16_t)); // (1024,128)
  bf16_t* WhT    = (bf16_t*)walloc((size_t)G_ * H_ * sizeof(bf16_t)); // (1024,256)
  bf16_t* W1T    = (bf16_t*)walloc((size_t)H_ * H_ * sizeof(bf16_t)); // (256,256)
  bf16_t* W2T    = (bf16_t*)walloc((size_t)(2 * Z_) * H_ * sizeof(bf16_t)); // (128,256)
  bf16_t* xb     = (bf16_t*)walloc((size_t)S_ * X_ * sizeof(bf16_t)); // 32 MB
  bf16_t* h_all  = (bf16_t*)walloc((size_t)S_ * H_ * sizeof(bf16_t)); // 64 MB
  bf16_t* hid    = (bf16_t*)walloc((size_t)S_ * H_ * sizeof(bf16_t)); // 64 MB
  float*  x_proj = (float*) walloc((size_t)S_ * G_ * sizeof(float));  // 512 MB

  // 1) weights -> bf16 transposed (N x K)
  transpose_to_bf16_kernel<<<(X_ * G_ + 255) / 256, 256, 0, stream>>>(Wi, WiT, X_, G_);
  transpose_to_bf16_kernel<<<(H_ * G_ + 255) / 256, 256, 0, stream>>>(Wh, WhT, H_, G_);
  transpose_to_bf16_kernel<<<(H_ * H_ + 255) / 256, 256, 0, stream>>>(W1, W1T, H_, H_);
  transpose_to_bf16_kernel<<<(H_ * 2 * Z_ + 255) / 256, 256, 0, stream>>>(W2, W2T, H_, 2 * Z_);

  // 2) x -> bf16
  f32_to_bf16_kernel<<<4096, 256, 0, stream>>>(x, xb, (long long)S_ * X_);

  // 3) x_proj = x @ Wi   (M=131072, N=1024, K=128) -> fp32, (T,B,4H) layout
  {
    int waves = (S_ / 16) * (G_ / 64);
    wmma_gemm_kernel<0, G_, X_><<<waves / 8, 256, 0, stream>>>(xb, WiT, nullptr,
                                                               x_proj, nullptr);
  }

  // 4) serial scan: 4 independent batch blocks, c in registers, h in LDS
  lstm_scan_kernel<<<4, 512, 0, stream>>>(x_proj, WhT, bh, h_all);

  // 5) hid = relu(h_all @ W1 + b1)  (M=131072, N=256, K=256) -> bf16
  {
    int waves = (S_ / 16) * (H_ / 64);
    wmma_gemm_kernel<1, H_, H_><<<waves / 8, 256, 0, stream>>>(h_all, W1T, b1,
                                                               nullptr, hid);
  }

  // 6) out = hid @ W2 + b2, split mu / log_sigma  (M=131072, N=128, K=256)
  {
    int waves = (S_ / 16) * ((2 * Z_) / 64);
    wmma_gemm_kernel<2, 2 * Z_, H_><<<waves / 8, 256, 0, stream>>>(hid, W2T, b2,
                                                                   out, nullptr);
  }
}
